// GatedFeatureLearningUnit_48704929136838
// MI455X (gfx1250) — matrix operand order repaired and checked by reading, compiled, not run on hardware
//
#include <hip/hip_runtime.h>

// ---------------------------------------------------------------------------
// GFLU (gated feature learning unit), F=512, 6 stages, batch 16384.
// Fully fused: each workgroup carries a 32-row batch tile through all 6
// stages in LDS; GEMMs via v_wmma_f32_16x16x32_bf16 (f32 accumulate).
// Register blocking: 2 M-tiles x 4 N-tiles per wave -> 8 WMMAs per pair of
// A-fragment LDS loads (0.5 ds_load_b128 per WMMA).
// ---------------------------------------------------------------------------

#define F_IN     512
#define F2       1024
#define N_STAGES 6
#define BM       32          // batch rows per workgroup
#define THREADS  256         // 8 waves (wave32)
#define APITCH   1032        // ushort pitch for bf16 LDS tiles (bank-conflict pad)

typedef __attribute__((ext_vector_type(16))) __bf16         v16bf;
typedef __attribute__((ext_vector_type(8)))  float          v8f;
typedef __attribute__((ext_vector_type(8)))  unsigned short v8u;

union FragU { v8u h[2]; v16bf b; };

__device__ __forceinline__ unsigned short f2bf(float f) {
  unsigned int u = __float_as_uint(f);
  u += 0x7FFFu + ((u >> 16) & 1u);            // round to nearest even
  return (unsigned short)(u >> 16);
}
__device__ __forceinline__ float bf2f(unsigned short h) {
  return __uint_as_float(((unsigned int)h) << 16);
}
__device__ __forceinline__ float sigmoidf_(float t) {
  return 1.0f / (1.0f + __expf(-t));
}
__device__ __forceinline__ float tanhf_(float t) {
  float e = __expf(-2.0f * t);
  return (1.0f - e) / (1.0f + e);
}

// ---------------------------------------------------------------------------
// Kernel 1: convert f32 weights -> bf16 workspace (Win then Wout, flat).
// ---------------------------------------------------------------------------
__global__ void cvt_kernel(const float* __restrict__ Win_w,
                           const float* __restrict__ Wout_w,
                           unsigned short* __restrict__ wbf,
                           int nWin, int nTotal) {
  int stride = gridDim.x * blockDim.x;
  for (int i = blockIdx.x * blockDim.x + threadIdx.x; i < nTotal; i += stride) {
    float v = (i < nWin) ? Win_w[i] : Wout_w[i - nWin];
    wbf[i] = f2bf(v);
  }
}

// ---------------------------------------------------------------------------
// Kernel 2: exact entmax-1.5 on each 512-wide mask row (one block per stage).
// Bitonic sort + Hillis-Steele scans in LDS.
// ---------------------------------------------------------------------------
__global__ __launch_bounds__(512)
void entmax_kernel(const float* __restrict__ masks, float* __restrict__ out) {
  __shared__ float s_x[512];    // sorted ascending
  __shared__ float s_a[512];    // cumsum of descending xs
  __shared__ float s_b[512];    // cumsum of descending xs^2
  __shared__ float s_tau[512];
  __shared__ float s_red[512];

  const int d = blockIdx.x;
  const int t = threadIdx.x;

  float v = masks[d * F_IN + t];
  s_red[t] = v; __syncthreads();
  for (int off = 256; off > 0; off >>= 1) {
    if (t < off) s_red[t] = fmaxf(s_red[t], s_red[t + off]);
    __syncthreads();
  }
  const float xv = (v - s_red[0]) * 0.5f;
  __syncthreads();

  s_x[t] = xv; __syncthreads();
  // bitonic sort ascending
  for (int k = 2; k <= 512; k <<= 1) {
    for (int j = k >> 1; j > 0; j >>= 1) {
      int ixj = t ^ j;
      if (ixj > t) {
        float a = s_x[t], b = s_x[ixj];
        bool asc = ((t & k) == 0);
        if ((a > b) == asc) { s_x[t] = b; s_x[ixj] = a; }
      }
      __syncthreads();
    }
  }
  const float xd = s_x[511 - t];              // descending view
  s_a[t] = xd; s_b[t] = xd * xd; __syncthreads();
  for (int off = 1; off < 512; off <<= 1) {
    float av = s_a[t], bv = s_b[t];
    float ap = (t >= off) ? s_a[t - off] : 0.0f;
    float bp = (t >= off) ? s_b[t - off] : 0.0f;
    __syncthreads();
    s_a[t] = av + ap; s_b[t] = bv + bp;
    __syncthreads();
  }
  const float rho  = (float)(t + 1);
  const float mean = s_a[t] / rho;
  const float msq  = s_b[t] / rho;
  const float ss   = rho * (msq - mean * mean);
  const float delta = fmaxf((1.0f - ss) / rho, 0.0f);
  const float tau  = mean - sqrtf(delta);
  s_tau[t] = tau;
  s_red[t] = (tau <= xd) ? 1.0f : 0.0f;
  __syncthreads();
  for (int off = 256; off > 0; off >>= 1) {
    if (t < off) s_red[t] += s_red[t + off];
    __syncthreads();
  }
  const int support = (int)(s_red[0] + 0.5f);
  const float tau_star = s_tau[support - 1];
  float p = fmaxf(xv - tau_star, 0.0f);
  out[d * F_IN + t] = p * p;
}

// ---------------------------------------------------------------------------
// Kernel 3: fused 6-stage GFLU. 8 waves; WMMA bf16 GEMMs with f32 accum.
// ---------------------------------------------------------------------------
__global__ __launch_bounds__(THREADS, 1)
void gflu_kernel(const float* __restrict__ x,
                 const float* __restrict__ Win_b,
                 const float* __restrict__ Wout_b,
                 const unsigned short* __restrict__ Win_bf,
                 const unsigned short* __restrict__ Wout_bf,
                 const float* __restrict__ scales,
                 float* __restrict__ out) {
  extern __shared__ char smem[];
  float*          hbuf = (float*)smem;                                  // BM*512 f32
  unsigned short* Abuf = (unsigned short*)(smem + BM * F_IN * 4);       // BM*APITCH bf16
  unsigned short* gbuf = Abuf + BM * APITCH;                            // BM*APITCH bf16

  const int tid  = threadIdx.x;
  const int wave = tid >> 5;
  const int lane = tid & 31;
  const int m    = lane & 15;
  const int hi   = lane >> 4;          // half-wave select
  const int klo  = hi ? 8 : 0;         // A-fragment K base (ISA 16-bit A 16x32)
  const int kbB  = hi ? 8 : 0;         // B-fragment K base (ISA B 32x16)
  const int row0 = blockIdx.x * BM;

  // h = x
  for (int i = tid; i < BM * F_IN; i += THREADS) {
    int r = i >> 9, c = i & (F_IN - 1);
    hbuf[i] = x[(size_t)(row0 + r) * F_IN + c];
  }
  __syncthreads();

  for (int d = 0; d < N_STAGES; ++d) {
    const float* sc = scales + d * F_IN;

    // ---- A1 = [entmax(mask)*x , h]  (bf16) ----
    for (int i = tid; i < BM * F2; i += THREADS) {
      int r = i >> 10, c = i & (F2 - 1);
      float v = (c < F_IN) ? sc[c] * x[(size_t)(row0 + r) * F_IN + c]
                           : hbuf[r * F_IN + (c - F_IN)];
      Abuf[r * APITCH + c] = f2bf(v);
    }
    __syncthreads();

    // ---- GEMM1: h_in = A1 @ Win^T + b ; g = sigmoid(h_in) -> gbuf ----
    {
      const unsigned short* W1 = Win_bf + (size_t)d * F2 * F2;
      const float* b1 = Win_b + d * F2;
      for (int ntg = 0; ntg < 2; ++ntg) {
        const int ntbase = wave * 8 + ntg * 4;     // 4 N-tiles per group
        const unsigned short* wrow[4];
#pragma unroll
        for (int j = 0; j < 4; ++j)
          wrow[j] = W1 + (size_t)((ntbase + j) * 16 + m) * F2 + kbB;
        v8f acc0[4] = {{}, {}, {}, {}};
        v8f acc1[4] = {{}, {}, {}, {}};
        for (int kt = 0; kt < F2 / 32; ++kt) {
          FragU a0_, a1_;
          const unsigned short* ap0 = Abuf + m * APITCH + kt * 32 + klo;
          const unsigned short* ap1 = ap0 + 16 * APITCH;
          a0_.h[0] = *(const v8u*)(ap0);
          a0_.h[1] = *(const v8u*)(ap0 + 16);
          a1_.h[0] = *(const v8u*)(ap1);
          a1_.h[1] = *(const v8u*)(ap1 + 16);
#pragma unroll
          for (int j = 0; j < 4; ++j) {
            FragU bf_;
            bf_.h[0] = *(const v8u*)(wrow[j] + kt * 32);
            bf_.h[1] = *(const v8u*)(wrow[j] + kt * 32 + 16);
            acc0[j] = __builtin_amdgcn_wmma_f32_16x16x32_bf16(false, a0_.b, false, bf_.b,
                                                              (short)0, acc0[j], false, false);
            acc1[j] = __builtin_amdgcn_wmma_f32_16x16x32_bf16(false, a1_.b, false, bf_.b,
                                                              (short)0, acc1[j], false, false);
          }
        }
#pragma unroll
        for (int j = 0; j < 4; ++j) {
          const int ncol = (ntbase + j) * 16 + m;  // 0..1023
          const float bias = b1[ncol];
#pragma unroll
          for (int rr = 0; rr < 8; ++rr) {
            const int mr = rr + (hi ? 8 : 0);
            gbuf[mr * APITCH + ncol]        = f2bf(sigmoidf_(acc0[j][rr] + bias));
            gbuf[(16 + mr) * APITCH + ncol] = f2bf(sigmoidf_(acc1[j][rr] + bias));
          }
        }
      }
    }
    __syncthreads();

    // ---- A2 = [r*h , x]  (bf16), r = gbuf[:,512:] ----
    for (int i = tid; i < BM * F2; i += THREADS) {
      int r = i >> 10, c = i & (F2 - 1);
      float v = (c < F_IN)
                  ? bf2f(gbuf[r * APITCH + F_IN + c]) * hbuf[r * F_IN + c]
                  : x[(size_t)(row0 + r) * F_IN + (c - F_IN)];
      Abuf[r * APITCH + c] = f2bf(v);
    }
    __syncthreads();

    // ---- GEMM2: h_out = tanh(A2 @ Wout^T + b); h = (1-z)h + z*h_out ----
    {
      const unsigned short* W2 = Wout_bf + (size_t)d * F_IN * F2;
      const float* b2 = Wout_b + d * F_IN;
      const int ntbase = wave * 4;                 // 4 N-tiles per wave
      const unsigned short* wrow[4];
#pragma unroll
      for (int j = 0; j < 4; ++j)
        wrow[j] = W2 + (size_t)((ntbase + j) * 16 + m) * F2 + kbB;
      v8f acc0[4] = {{}, {}, {}, {}};
      v8f acc1[4] = {{}, {}, {}, {}};
      for (int kt = 0; kt < F2 / 32; ++kt) {
        FragU a0_, a1_;
        const unsigned short* ap0 = Abuf + m * APITCH + kt * 32 + klo;
        const unsigned short* ap1 = ap0 + 16 * APITCH;
        a0_.h[0] = *(const v8u*)(ap0);
        a0_.h[1] = *(const v8u*)(ap0 + 16);
        a1_.h[0] = *(const v8u*)(ap1);
        a1_.h[1] = *(const v8u*)(ap1 + 16);
#pragma unroll
        for (int j = 0; j < 4; ++j) {
          FragU bf_;
          bf_.h[0] = *(const v8u*)(wrow[j] + kt * 32);
          bf_.h[1] = *(const v8u*)(wrow[j] + kt * 32 + 16);
          acc0[j] = __builtin_amdgcn_wmma_f32_16x16x32_bf16(false, a0_.b, false, bf_.b,
                                                            (short)0, acc0[j], false, false);
          acc1[j] = __builtin_amdgcn_wmma_f32_16x16x32_bf16(false, a1_.b, false, bf_.b,
                                                            (short)0, acc1[j], false, false);
        }
      }
#pragma unroll
      for (int j = 0; j < 4; ++j) {
        const int ncol = (ntbase + j) * 16 + m;    // 0..511
        const float bias = b2[ncol];
#pragma unroll
        for (int rr = 0; rr < 8; ++rr) {
          const int mr = rr + (hi ? 8 : 0);
          {
            float hout = tanhf_(acc0[j][rr] + bias);
            float z = bf2f(gbuf[mr * APITCH + ncol]);
            float h = hbuf[mr * F_IN + ncol];
            hbuf[mr * F_IN + ncol] = (1.0f - z) * h + z * hout;
          }
          {
            float hout = tanhf_(acc1[j][rr] + bias);
            float z = bf2f(gbuf[(16 + mr) * APITCH + ncol]);
            float h = hbuf[(16 + mr) * F_IN + ncol];
            hbuf[(16 + mr) * F_IN + ncol] = (1.0f - z) * h + z * hout;
          }
        }
      }
    }
    __syncthreads();
  }

  // write h
  for (int i = tid; i < BM * F_IN; i += THREADS) {
    int r = i >> 9, c = i & (F_IN - 1);
    out[(size_t)(row0 + r) * F_IN + c] = hbuf[i];
  }
}

// ---------------------------------------------------------------------------
// Launch. Inputs: x, masks, Win_w, Win_b, Wout_w, Wout_b.
// Workspace layout (bytes):
//   [0, 12582912)            Win  bf16 (6*1024*1024)
//   [12582912, 18874368)     Wout bf16 (6*512*1024)
//   [18874368, 18886656)     entmax scales f32 (6*512)
// ---------------------------------------------------------------------------
extern "C" void kernel_launch(void* const* d_in, const int* in_sizes, int n_in,
                              void* d_out, int out_size, void* d_ws, size_t ws_size,
                              hipStream_t stream) {
  const float* x      = (const float*)d_in[0];
  const float* masks  = (const float*)d_in[1];
  const float* Win_w  = (const float*)d_in[2];
  const float* Win_b  = (const float*)d_in[3];
  const float* Wout_w = (const float*)d_in[4];
  const float* Wout_b = (const float*)d_in[5];

  const int nWin   = N_STAGES * F2 * F2;       // 6,291,456
  const int nWout  = N_STAGES * F_IN * F2;     // 3,145,728
  const int nTotal = nWin + nWout;

  unsigned short* wbf = (unsigned short*)d_ws;
  float* scales = (float*)((char*)d_ws + (size_t)nTotal * sizeof(unsigned short));

  cvt_kernel<<<(nTotal + 255) / 256, 256, 0, stream>>>(Win_w, Wout_w, wbf, nWin, nTotal);
  entmax_kernel<<<N_STAGES, 512, 0, stream>>>(masks, scales);

  const size_t smem_bytes = (size_t)BM * F_IN * 4         // hbuf
                          + (size_t)BM * APITCH * 2       // Abuf
                          + (size_t)BM * APITCH * 2;      // gbuf  ~193 KB
  gflu_kernel<<<16384 / BM, THREADS, smem_bytes, stream>>>(
      x, Win_b, Wout_b, wbf, wbf + nWin, scales, (float*)d_out);
}
